// AutoregressiveDecoderLayer_38448547234765
// MI455X (gfx1250) — matrix operand order repaired and checked by reading, not compiled, not run on hardware
//
#include <hip/hip_runtime.h>
#include <hip/hip_bf16.h>

#define B_  256
#define T_  512
#define F_  128
#define CU_ 512
#define DU_ 512
#define KTOT 1152           // F_ + CU_ + DU_
#define NCOL 2048           // 4*DU_
#define NBLK 256            // persistent grid blocks
#define XSTR 1160           // padded LDS row stride (elements): rows hit distinct banks

typedef __attribute__((ext_vector_type(16))) __bf16 bf16x16;
typedef __attribute__((ext_vector_type(8)))  __bf16 bf16x8;
typedef __attribute__((ext_vector_type(4)))  __bf16 bf16x4;
typedef __attribute__((ext_vector_type(8)))  float  f32x8;

__device__ __forceinline__ float sigf(float x) { return 1.0f / (1.0f + __expf(-x)); }

__device__ __forceinline__ bf16x4 cvt4(float4 v) {
  bf16x4 s;
  s[0] = (__bf16)v.x; s[1] = (__bf16)v.y; s[2] = (__bf16)v.z; s[3] = (__bf16)v.w;
  return s;
}

// ---------------------------------------------------------------------------
// Operand loaders. CDNA5 16-bit A layout: lane (m = lane&15) holds K-chunks
// [k0+half*8, +8) and [k0+16+half*8, +8).  B layout (from transposed weights):
// lane (n = lane&15) holds K-chunks [k0+half*8, +8) and [k0+16+half*8, +8).
// ---------------------------------------------------------------------------
__device__ __forceinline__ bf16x16 load_a_bf16(const __bf16* p, int half) {
  bf16x8 lo = *(const bf16x8*)(p + half * 8);             // b128
  bf16x8 hi = *(const bf16x8*)(p + 16 + half * 8);        // b128
  bf16x16 a;
#pragma unroll
  for (int e = 0; e < 8; ++e) { a[e] = lo[e]; a[8 + e] = hi[e]; }
  return a;
}

__device__ __forceinline__ bf16x16 load_b_tr(const __bf16* p, int half) {
  // p = &WT[col*K + k0]
  bf16x8 lo = *(const bf16x8*)(p + half * 8);             // b128
  bf16x8 hi = *(const bf16x8*)(p + 16 + half * 8);        // b128
  bf16x16 b;
#pragma unroll
  for (int e = 0; e < 8; ++e) { b[e] = lo[e]; b[8 + e] = hi[e]; }
  return b;
}

// ---------------------------------------------------------------------------
// Device-wide barrier: monotonic-generation counter barrier.
// ---------------------------------------------------------------------------
__device__ __forceinline__ void grid_sync(unsigned* cnt, unsigned* gen, unsigned nblk) {
  __syncthreads();
  if (threadIdx.x == 0) {
    __threadfence();
    unsigned g = atomicAdd(gen, 0u);
    unsigned arrived = atomicAdd(cnt, 1u);
    if (arrived == nblk - 1u) {
      atomicExch(cnt, 0u);
      __threadfence();
      atomicAdd(gen, 1u);
    } else {
      while (atomicAdd(gen, 0u) == g) { __builtin_amdgcn_s_sleep(1); }
    }
    __threadfence();
  }
  __syncthreads();
}

// ---------------------------------------------------------------------------
// K0: pack transposed bf16 weights, init h/c state, reset barrier counters.
//   WUT[j][k] = (k<640 ? W[k][j] : U[k-640][j])   j<2048, k<1152
//   WoT[f][k] = Wo[k][f]                          f<128,  k<512
// ---------------------------------------------------------------------------
__global__ void pack_init(const float* __restrict__ W, const float* __restrict__ U,
                          const float* __restrict__ Wo,
                          const float* __restrict__ h0, const float* __restrict__ c0,
                          __bf16* __restrict__ WUT, __bf16* __restrict__ WoT,
                          __bf16* __restrict__ h_bf, float* __restrict__ h_f,
                          float* __restrict__ c_f, unsigned* __restrict__ bar) {
  const unsigned idx0 = blockIdx.x * blockDim.x + threadIdx.x;
  const unsigned stride = gridDim.x * blockDim.x;
  for (unsigned i = idx0; i < (unsigned)KTOT * NCOL; i += stride) {
    unsigned j = i & (NCOL - 1), k = i >> 11;            // coalesced source read
    float v = (k < (F_ + CU_)) ? W[k * NCOL + j] : U[(k - (F_ + CU_)) * NCOL + j];
    WUT[(size_t)j * KTOT + k] = (__bf16)v;
  }
  for (unsigned i = idx0; i < (unsigned)DU_ * F_; i += stride) {
    unsigned f = i & (F_ - 1), k = i >> 7;
    WoT[(size_t)f * DU_ + k] = (__bf16)Wo[k * F_ + f];
  }
  for (unsigned i = idx0; i < (unsigned)B_ * DU_; i += stride) {
    h_bf[i] = (__bf16)h0[i];
    h_f[i]  = h0[i];
    c_f[i]  = c0[i];
  }
  if (idx0 < 2) bar[idx0] = 0u;
}

// ---------------------------------------------------------------------------
// K1: persistent LSTM. Block = 16 batch rows x 128 gate cols; its 8 waves each
// own one 16x16 z tile. Per step: stage x_t = [prev|conductor|h] (bf16) for the
// block's 16 rows in LDS once (8x fewer L2 A-reads + converts), then a uniform
// 36-step K loop: 2x ds_load_b128 (A) + 2x global_load_b128 (B) + WMMA, with
// two interleaved accumulators. Then grid barrier, fused gates, barrier.
// ---------------------------------------------------------------------------
__global__ void lstm_persistent(const float* __restrict__ teacher,
                                const float* __restrict__ conductor,
                                const float* __restrict__ bias,
                                const __bf16* __restrict__ WUT,
                                __bf16* __restrict__ h_bf,
                                float* __restrict__ h_f,
                                float* __restrict__ c_f,
                                float* __restrict__ z,
                                __bf16* __restrict__ hs,
                                unsigned* __restrict__ bar) {
  __shared__ __align__(16) __bf16 xs[16 * XSTR];          // 37.1 KB

  const int wid   = blockIdx.x * 8 + (threadIdx.x >> 5);   // 0..2047: tile id
  const int lane  = threadIdx.x & 31;
  const int lan   = lane & 15;
  const int half  = lane >> 4;
  const int mtile = wid >> 7;            // 16 M-tiles (batch) — same for whole block
  const int ntile = wid & 127;           // 128 N-tiles (gate cols)
  const int m0 = mtile * 16, n0 = ntile * 16;
  const int tb  = threadIdx.x;
  const int tid = blockIdx.x * 256 + threadIdx.x;
  const __bf16* pB = WUT + (size_t)(n0 + lan) * KTOT;      // this lane's B column

  for (int t = 0; t < T_; ++t) {
    // ================= stage x_t slab for rows [m0, m0+16) =================
    // teacher region: k in [0,128): 512 quads over 256 threads
    if (t > 0) {
#pragma unroll
      for (int it = 0; it < 2; ++it) {
        const int idx = tb + it * 256;
        const int r = idx >> 5, q = idx & 31;
        float4 v = *(const float4*)(teacher + ((size_t)(m0 + r) * T_ + (t - 1)) * F_ + q * 4);
        *(bf16x4*)(xs + r * XSTR + q * 4) = cvt4(v);
      }
    } else {
      const bf16x4 zz = (bf16x4){(__bf16)0.f, (__bf16)0.f, (__bf16)0.f, (__bf16)0.f};
#pragma unroll
      for (int it = 0; it < 2; ++it) {
        const int idx = tb + it * 256;
        const int r = idx >> 5, q = idx & 31;
        *(bf16x4*)(xs + r * XSTR + q * 4) = zz;
      }
    }
    // conductor region: k in [128,640): 2048 quads
#pragma unroll
    for (int it = 0; it < 8; ++it) {
      const int idx = tb + it * 256;
      const int r = idx >> 7, q = idx & 127;
      float4 v = *(const float4*)(conductor + ((size_t)(m0 + r) * T_ + t) * CU_ + q * 4);
      *(bf16x4*)(xs + r * XSTR + F_ + q * 4) = cvt4(v);
    }
    // recurrent region: k in [640,1152): 1024 bf16x8 chunks
#pragma unroll
    for (int it = 0; it < 4; ++it) {
      const int idx = tb + it * 256;
      const int r = idx >> 6, q = idx & 63;
      bf16x8 v = *(const bf16x8*)(h_bf + (size_t)(m0 + r) * DU_ + q * 8);
      *(bf16x8*)(xs + r * XSTR + (F_ + CU_) + q * 8) = v;
    }
    __syncthreads();

    // ======================= z tile GEMM, K = 1152 ==========================
    f32x8 acc0 = (f32x8){0.f, 0.f, 0.f, 0.f, 0.f, 0.f, 0.f, 0.f};
    f32x8 acc1 = acc0;
    const __bf16* pa = xs + lan * XSTR;                   // A row in LDS
#pragma unroll 2
    for (int ks = 0; ks < KTOT / 32; ks += 2) {
      bf16x16 a0 = load_a_bf16(pa + ks * 32, half);
      bf16x16 a1 = load_a_bf16(pa + ks * 32 + 32, half);
      bf16x16 b0 = load_b_tr(pB + ks * 32, half);
      bf16x16 b1 = load_b_tr(pB + ks * 32 + 32, half);
      __builtin_prefetch(pB + ks * 32 + 256, 0, 1);
      acc0 = __builtin_amdgcn_wmma_f32_16x16x32_bf16(false, a0, false, b0, (short)0, acc0, false, false);
      acc1 = __builtin_amdgcn_wmma_f32_16x16x32_bf16(false, a1, false, b1, (short)0, acc1, false, false);
    }

    // D layout: VGPR r -> (M = r + 8*half, N = lane&15)
#pragma unroll
    for (int r = 0; r < 8; ++r)
      z[(m0 + r + half * 8) * NCOL + n0 + lan] = acc0[r] + acc1[r];

    grid_sync(bar, bar + 1, NBLK);

    // ---- fused gates: 131072 elements over 65536 threads ----
#pragma unroll
    for (int rep = 0; rep < 2; ++rep) {
      const int i  = tid + rep * (NBLK * 256);
      const int bb = i >> 9;
      const int d  = i & (DU_ - 1);
      const float* zr = z + bb * NCOL;
      const float iv = zr[d]            + bias[d];
      const float fv = zr[DU_ + d]      + bias[DU_ + d];
      const float gv = zr[2 * DU_ + d]  + bias[2 * DU_ + d];
      const float ov = zr[3 * DU_ + d]  + bias[3 * DU_ + d];
      const float cn = sigf(fv) * c_f[i] + sigf(iv) * tanhf(gv);
      const float hn = sigf(ov) * tanhf(cn);
      c_f[i] = cn;
      h_f[i] = hn;
      const __bf16 hb = (__bf16)hn;
      h_bf[i] = hb;
      hs[(size_t)t * (B_ * DU_) + i] = hb;
    }

    grid_sync(bar, bar + 1, NBLK);
  }
}

// ---------------------------------------------------------------------------
// K2: out = sigmoid(hs @ Wo + bo). [131072 x 512] x [512 x 128].
// One wave per 16-row M-tile; 8 independent accumulator chains (A reuse).
// ---------------------------------------------------------------------------
__global__ void out_gemm(const __bf16* __restrict__ hs,
                         const __bf16* __restrict__ WoT,
                         const float* __restrict__ bo,
                         float* __restrict__ out) {
  const int wid  = blockIdx.x * 8 + (threadIdx.x >> 5);  // 0..8191 M-tile
  const int lane = threadIdx.x & 31;
  const int lan  = lane & 15;
  const int half = lane >> 4;
  const int row  = wid * 16 + lan;

  f32x8 acc[8];
#pragma unroll
  for (int nt = 0; nt < 8; ++nt) acc[nt] = (f32x8){0.f, 0.f, 0.f, 0.f, 0.f, 0.f, 0.f, 0.f};

  const __bf16* pa = hs + (size_t)row * DU_;
  for (int ks = 0; ks < DU_ / 32; ++ks) {
    bf16x16 a = load_a_bf16(pa + ks * 32, half);
#pragma unroll
    for (int nt = 0; nt < 8; ++nt) {
      bf16x16 bm = load_b_tr(WoT + (size_t)(nt * 16 + lan) * DU_ + ks * 32, half);
      acc[nt] = __builtin_amdgcn_wmma_f32_16x16x32_bf16(false, a, false, bm,
                                                        (short)0, acc[nt], false, false);
    }
  }

  const int mbase = wid * 16;
#pragma unroll
  for (int nt = 0; nt < 8; ++nt) {
    const int f0 = nt * 16 + lan;
    const float bv = bo[f0];
#pragma unroll
    for (int r = 0; r < 8; ++r) {
      const int grow = mbase + r + half * 8;    // global row = t*B + b
      const int t  = grow >> 8;
      const int bb = grow & (B_ - 1);
      const float v = acc[nt][r] + bv;
      out[((size_t)bb * T_ + t) * F_ + f0] = sigf(v);
    }
  }
}

// ---------------------------------------------------------------------------
// K3: hT / cT tail of d_out.
// ---------------------------------------------------------------------------
__global__ void finalize(const float* __restrict__ h_f, const float* __restrict__ c_f,
                         float* __restrict__ dst) {
  const int i = blockIdx.x * blockDim.x + threadIdx.x;  // 131072 threads
  dst[i] = h_f[i];
  dst[B_ * DU_ + i] = c_f[i];
}

// ---------------------------------------------------------------------------
extern "C" void kernel_launch(void* const* d_in, const int* in_sizes, int n_in,
                              void* d_out, int out_size, void* d_ws, size_t ws_size,
                              hipStream_t stream) {
  (void)in_sizes; (void)n_in; (void)out_size; (void)ws_size;

  const float* conductor = (const float*)d_in[0];  // [B,T,CU]
  const float* teacher   = (const float*)d_in[1];  // [B,T,F]
  const float* h0        = (const float*)d_in[2];  // [B,DU]
  const float* c0        = (const float*)d_in[3];  // [B,DU]
  const float* W         = (const float*)d_in[4];  // [F+CU, 4DU]
  const float* U         = (const float*)d_in[5];  // [DU, 4DU]
  const float* b         = (const float*)d_in[6];  // [4DU]
  const float* Wo        = (const float*)d_in[7];  // [DU, F]
  const float* bo        = (const float*)d_in[8];  // [F]
  float* out = (float*)d_out;                      // out | hT | cT

  // ---- workspace partition (≈142 MB) ----
  char* ws = (char*)d_ws;
  unsigned* bar = (unsigned*)ws;                                 // 256 B
  __bf16* WUT  = (__bf16*)(ws + 256);                            // 2048*1152
  __bf16* WoT  = WUT + (size_t)NCOL * KTOT;                      // 128*512
  __bf16* h_bf = WoT + (size_t)F_ * DU_;                         // 256*512
  __bf16* hs   = h_bf + (size_t)B_ * DU_;                        // 512*256*512
  float*  h_f  = (float*)(hs + (size_t)T_ * B_ * DU_);
  float*  c_f  = h_f + (size_t)B_ * DU_;
  float*  z    = c_f + (size_t)B_ * DU_;                         // 256*2048

  pack_init<<<2048, 256, 0, stream>>>(W, U, Wo, h0, c0, WUT, WoT, h_bf, h_f, c_f, bar);
  lstm_persistent<<<NBLK, 256, 0, stream>>>(teacher, conductor, b, WUT,
                                            h_bf, h_f, c_f, z, hs, bar);
  out_gemm<<<(B_ * T_) / 16 / 8, 256, 0, stream>>>(hs, WoT, bo, out);
  finalize<<<(B_ * DU_) / 256, 256, 0, stream>>>(h_f, c_f, out + (size_t)B_ * T_ * F_);
}
